// PhysicsGuidedAttention_85040352460997
// MI455X (gfx1250) — matrix operand order repaired and checked by reading, not compiled, not run on hardware
//
#include <hip/hip_runtime.h>
#include <hip/hip_bf16.h>

// ---------------------------------------------------------------------------
// Physics-guided attention on gfx1250 (MI455X), bf16 WMMA + fp32 softmax,
// TDM (tensor_load_to_lds) staging of K tiles.
// B=2, N_ctx=4096, N_query=2048, C=512, H=8, D=64.
// ---------------------------------------------------------------------------

#define BATCH   2
#define NQ      2048
#define NC      4096
#define CEMB    512
#define NHEAD   8
#define HDIM    64

typedef __attribute__((ext_vector_type(16))) __bf16       v16bf;
typedef __attribute__((ext_vector_type(8)))  float        v8f;
typedef __attribute__((ext_vector_type(4)))  unsigned int v4u;
typedef __attribute__((ext_vector_type(8)))  unsigned int v8u;

union ABFrag { v16bf v; uint4 q[2]; };

__device__ __forceinline__ v8f wmma_bf16(v16bf a, v16bf b, v8f c) {
    // D = A(16x32 bf16) * B(32x16 bf16) + C(16x16 f32)
    return __builtin_amdgcn_wmma_f32_16x16x32_bf16(
        /*neg_a=*/false, a, /*neg_b=*/false, b,
        /*c_mod=*/(short)0, c, /*reuse_a=*/false, /*reuse_b=*/false);
}

// ---------------------------------------------------------------------------
// fp32 -> bf16 conversion (grid-stride)
// ---------------------------------------------------------------------------
__global__ void cvt_bf16_kernel(const float* __restrict__ src,
                                __bf16* __restrict__ dst, int n) {
    for (int i = blockIdx.x * blockDim.x + threadIdx.x; i < n;
         i += gridDim.x * blockDim.x)
        dst[i] = (__bf16)src[i];
}

// ---------------------------------------------------------------------------
// Projection GEMM: out = (A @ W^T + bias) * scale, written head-major bf16.
// A: [rowsTotal, 512] bf16 (rowsTotal = BATCH*Nper), W: [512,512] bf16 [out,in]
// out: [BATCH, NHEAD, Nper, HDIM] bf16
// One wave computes a 16(M) x 64(N) strip: A-frag reused across 4 WMMA tiles.
// ---------------------------------------------------------------------------
__global__ __launch_bounds__(256) void proj_qkv_kernel(
    const __bf16* __restrict__ A, const __bf16* __restrict__ W,
    const float* __restrict__ bias, __bf16* __restrict__ out,
    int rowsTotal, int Nper, float scale) {

    const int lane = threadIdx.x & 31;
    const int wave = threadIdx.x >> 5;
    const int strip = blockIdx.x * 8 + wave;
    const int nStrips = (rowsTotal >> 4) * 8;
    if (strip >= nStrips) return;

    const int mt = strip >> 3;
    const int nt = strip & 7;
    const int m0 = mt << 4;
    const int n0 = nt << 6;
    const int half = lane >> 4;     // 0: lanes 0-15, 1: lanes 16-31
    const int l16  = lane & 15;

    const __bf16* ap = A + (size_t)(m0 + l16) * CEMB;

    v8f acc[4] = {};
    for (int kb = 0; kb < CEMB; kb += 32) {
        ABFrag af;
        const int ko = kb + (half << 3);           // K = kb + 8*half + [0..7]
        af.q[0] = *(const uint4*)(ap + ko);
        af.q[1] = *(const uint4*)(ap + ko + 16);   // K = kb+16+8*half+[0..7]
#pragma unroll
        for (int t = 0; t < 4; ++t) {
            const int n = n0 + (t << 4) + l16;     // output channel (B col)
            const __bf16* wp = W + (size_t)n * CEMB + kb + (half << 3);
            ABFrag bf;
            bf.q[0] = *(const uint4*)(wp);
            bf.q[1] = *(const uint4*)(wp + 16);
            acc[t] = wmma_bf16(af.v, bf.v, acc[t]);
        }
    }

#pragma unroll
    for (int t = 0; t < 4; ++t) {
        const int n = n0 + (t << 4) + l16;
        const float bv = bias[n];
        const int h = n >> 6, d = n & 63;
#pragma unroll
        for (int e = 0; e < 8; ++e) {
            const int m = m0 + e + (half << 3);    // global row
            const int b = m / Nper;
            const int r = m - b * Nper;
            const float val = (acc[t][e] + bv) * scale;
            out[(((size_t)(b * NHEAD + h) * Nper + r) << 6) + d] = (__bf16)val;
        }
    }
}

// ---------------------------------------------------------------------------
// Output projection: f32 out = A @ Wo^T + bo.  A: [rows,512] bf16.
// ---------------------------------------------------------------------------
__global__ __launch_bounds__(256) void proj_out_kernel(
    const __bf16* __restrict__ A, const __bf16* __restrict__ W,
    const float* __restrict__ bias, float* __restrict__ out, int rowsTotal) {

    const int lane = threadIdx.x & 31;
    const int wave = threadIdx.x >> 5;
    const int strip = blockIdx.x * 8 + wave;
    const int nStrips = (rowsTotal >> 4) * 8;
    if (strip >= nStrips) return;

    const int m0 = (strip >> 3) << 4;
    const int n0 = (strip & 7) << 6;
    const int half = lane >> 4;
    const int l16  = lane & 15;

    const __bf16* ap = A + (size_t)(m0 + l16) * CEMB;

    v8f acc[4] = {};
    for (int kb = 0; kb < CEMB; kb += 32) {
        ABFrag af;
        const int ko = kb + (half << 3);
        af.q[0] = *(const uint4*)(ap + ko);
        af.q[1] = *(const uint4*)(ap + ko + 16);
#pragma unroll
        for (int t = 0; t < 4; ++t) {
            const int n = n0 + (t << 4) + l16;
            const __bf16* wp = W + (size_t)n * CEMB + kb + (half << 3);
            ABFrag bf;
            bf.q[0] = *(const uint4*)(wp);
            bf.q[1] = *(const uint4*)(wp + 16);
            acc[t] = wmma_bf16(af.v, bf.v, acc[t]);
        }
    }

#pragma unroll
    for (int t = 0; t < 4; ++t) {
        const int n = n0 + (t << 4) + l16;
        const float bv = bias[n];
#pragma unroll
        for (int e = 0; e < 8; ++e) {
            const int m = m0 + e + (half << 3);
            out[(size_t)m * CEMB + n] = acc[t][e] + bv;
        }
    }
}

// ---------------------------------------------------------------------------
// Flash attention with heat-kernel bias.
// Block = 128 threads (4 waves); one block per (b, h, 64-query tile).
// Wave w owns query rows [qtile*64 + w*16, +16).
// K tile staged into LDS by the Tensor Data Mover (tensor_load_to_lds),
// V tile transposed manually (TDM cannot transpose; P*V needs [d][j]).
// ---------------------------------------------------------------------------
__global__ __launch_bounds__(128) void attn_kernel(
    const __bf16* __restrict__ Qh,   // [B,H,NQ,64] bf16 (pre-scaled by 1/8)
    const __bf16* __restrict__ Kh,   // [B,H,NC,64] bf16
    const __bf16* __restrict__ Vh,   // [B,H,NC,64] bf16
    const float*  __restrict__ pos_ctx,    // [B,NC,3]
    const float*  __restrict__ pos_query,  // [B,NQ,3]
    __bf16* __restrict__ Y) {        // [B,NQ,512] bf16

    __shared__ __bf16 Kt[64 * 64];       // row-major [j][d]  (TDM destination)
    __shared__ __bf16 Vt[64 * 64];       // transposed [d][j]
    __shared__ __bf16 Pl[4][16 * 64];    // per-wave P tile, row-major
    __shared__ float  posc[64][4];       // ctx positions (x,y,t)

    const int bid  = blockIdx.x;
    const int qt   = bid & 31;                 // NQ/64 = 32
    const int h    = (bid >> 5) & 7;
    const int b    = bid >> 8;
    const int tid  = threadIdx.x;
    const int wave = tid >> 5;
    const int lane = tid & 31;
    const int half = lane >> 4;
    const int l16  = lane & 15;
    const int q0   = qt * 64 + wave * 16;      // first query row of this wave

    const __bf16* qp = Qh + (((size_t)(b * NHEAD + h) * NQ + q0) << 6);
    const __bf16* kp = Kh + (((size_t)(b * NHEAD + h) * NC) << 6);
    const __bf16* vp = Vh + (((size_t)(b * NHEAD + h) * NC) << 6);

    // LDS byte offset of Kt (flat LDS addresses map via low 32 bits).
    const unsigned int kt_lds = (unsigned int)(uintptr_t)(void*)&Kt[0];

    // Q fragments (A-layout), loaded once: K-dim = 64 -> two 32-chunks.
    ABFrag qa[2];
    {
        const __bf16* qrow = qp + ((size_t)l16 << 6);
#pragma unroll
        for (int kk = 0; kk < 2; ++kk) {
            const int ko = kk * 32 + (half << 3);
            qa[kk].q[0] = *(const uint4*)(qrow + ko);
            qa[kk].q[1] = *(const uint4*)(qrow + ko + 16);
        }
    }

    // Query positions for the 8 rows this lane accumulates.
    float qpx[8], qpy[8], qpt[8];
#pragma unroll
    for (int e = 0; e < 8; ++e) {
        const int r = q0 + e + (half << 3);
        const float* pq = pos_query + ((size_t)b * NQ + r) * 3;
        qpx[e] = pq[0]; qpy[e] = pq[1]; qpt[e] = pq[2];
    }

    float mrow[8], lrow[8];
    v8f o[4] = {};
#pragma unroll
    for (int e = 0; e < 8; ++e) { mrow[e] = -1e30f; lrow[e] = 0.0f; }

    for (int ct = 0; ct < NC / 64; ++ct) {
        __syncthreads();   // protect Kt/Vt from previous iteration's readers

        // --- TDM: async K tile (contiguous 64x64 bf16 = 8KB) -> LDS ---------
        // D# per CDNA5 ISA: 2D tensor, 2-byte
        // elements, tile 64x64, dim0_stride=64. Groups 2/3 omitted (NULL).
        if (wave == 0) {
            const unsigned long long ga =
                (unsigned long long)(uintptr_t)(kp + ((size_t)ct << 12));
            v4u g0;
            g0[0] = 1u;                                   // count=1 (valid D#)
            g0[1] = kt_lds;                               // lds_addr
            g0[2] = (unsigned int)ga;                     // global_addr[31:0]
            g0[3] = (unsigned int)((ga >> 32) & 0x1FFFFFFu)
                  | (2u << 30);                           // addr[56:32], type=2
            v8u g1;
            g1[0] = 1u << 16;        // workgroup_mask=0, data_size=1 (2 bytes)
            g1[1] = 64u << 16;       // tensor_dim0 = 64          (bits 79:48)
            g1[2] = 4096u << 16;     // tensor_dim1 = 4096 (lo16) (bits 111:80)
            g1[3] = 64u << 16;       // tensor_dim1 hi16=0, tile_dim0 = 64
            g1[4] = 64u;             // tile_dim1 = 64, tile_dim2 = 0
            g1[5] = 64u;             // tensor_dim0_stride = 64 (lo32)
            g1[6] = 0u;              // stride hi16 / tensor_dim1_stride lo16
            g1[7] = 0u;              // tensor_dim1_stride hi32
            asm volatile("tensor_load_to_lds %0, %1"
                         :: "s"(g0), "s"(g1)
                         : "memory");
        }

        // --- cooperative stage: V tile transposed, ctx positions ------------
        const uint4* vsrc = (const uint4*)(vp + ((size_t)ct << 12));
#pragma unroll
        for (int i = 0; i < 4; ++i) {
            const int idx = tid + i * 128;        // uint4 index into 64x64 tile
            const uint4 val = vsrc[idx];
            const int j  = idx >> 3;
            const int d0 = (idx & 7) << 3;
            const __bf16* pv = (const __bf16*)&val;
#pragma unroll
            for (int u = 0; u < 8; ++u) Vt[(d0 + u) * 64 + j] = pv[u];
        }
        for (int idx = tid; idx < 192; idx += 128)
            posc[idx / 3][idx % 3] =
                pos_ctx[((size_t)b * NC + ct * 64) * 3 + idx];

        if (ct + 1 < NC / 64)
            __builtin_prefetch(vp + ((size_t)(ct + 1) << 12), 0, 1);

        // Wait for the TDM transfer, then make Kt visible to all waves.
        if (wave == 0) __builtin_amdgcn_s_wait_tensorcnt(0);
        __syncthreads();

        // --- S = Q * K^T (4 accumulator tiles, 16x64 total) ---
        v8f s[4];
#pragma unroll
        for (int t = 0; t < 4; ++t) {
            v8f accs = {};
#pragma unroll
            for (int kk = 0; kk < 2; ++kk) {
                const __bf16* kr = &Kt[((t << 4) + l16) * 64 + kk * 32 + (half << 3)];
                ABFrag bf;
                bf.q[0] = *(const uint4*)(kr);
                bf.q[1] = *(const uint4*)(kr + 16);
                accs = wmma_bf16(qa[kk].v, bf.v, accs);
            }
            s[t] = accs;
        }

        // --- heat-kernel bias + per-row tile max ---
        float tmax[8];
#pragma unroll
        for (int e = 0; e < 8; ++e) tmax[e] = -1e30f;
#pragma unroll
        for (int t = 0; t < 4; ++t) {
            const int j = (t << 4) + l16;
            const float cx = posc[j][0], cy = posc[j][1], cti = posc[j][2];
#pragma unroll
            for (int e = 0; e < 8; ++e) {
                const float dx = qpx[e] - cx;
                const float dy = qpy[e] - cy;
                const float dt = qpt[e] - cti;
                const float denom = 0.4f * fabsf(dt) + 1e-6f; // 4*nu*|dt|+eps
                const float heat = __expf(-(dx * dx + dy * dy) / denom);
                const float val = s[t][e] + heat;   // q pre-scaled by 1/8
                s[t][e] = val;
                tmax[e] = fmaxf(tmax[e], val);
            }
        }
        // butterfly max across the 16 lanes holding this row's columns
#pragma unroll
        for (int off = 1; off < 16; off <<= 1)
#pragma unroll
            for (int e = 0; e < 8; ++e)
                tmax[e] = fmaxf(tmax[e], __shfl_xor(tmax[e], off, 32));

        // --- online softmax update ---
        float corr[8], tsum[8];
#pragma unroll
        for (int e = 0; e < 8; ++e) {
            const float mn = fmaxf(mrow[e], tmax[e]);
            corr[e] = __expf(mrow[e] - mn);
            mrow[e] = mn;
            tsum[e] = 0.0f;
        }
#pragma unroll
        for (int t = 0; t < 4; ++t)
#pragma unroll
            for (int e = 0; e < 8; ++e) {
                const float p = __expf(s[t][e] - mrow[e]);
                s[t][e] = p;
                tsum[e] += p;
            }
#pragma unroll
        for (int off = 1; off < 16; off <<= 1)
#pragma unroll
            for (int e = 0; e < 8; ++e)
                tsum[e] += __shfl_xor(tsum[e], off, 32);
#pragma unroll
        for (int e = 0; e < 8; ++e) lrow[e] = lrow[e] * corr[e] + tsum[e];
#pragma unroll
        for (int t = 0; t < 4; ++t)
#pragma unroll
            for (int e = 0; e < 8; ++e) o[t][e] *= corr[e];

        // --- P (D-layout) -> LDS -> A-layout for P*V ---
        __bf16* pw = &Pl[wave][0];
#pragma unroll
        for (int t = 0; t < 4; ++t)
#pragma unroll
            for (int e = 0; e < 8; ++e)
                pw[(e + (half << 3)) * 64 + (t << 4) + l16] = (__bf16)s[t][e];
        __syncthreads();

        ABFrag pa[2];
        {
            const __bf16* pr = pw + l16 * 64;
#pragma unroll
            for (int kk = 0; kk < 2; ++kk) {
                const int ko = kk * 32 + (half << 3);
                pa[kk].q[0] = *(const uint4*)(pr + ko);
                pa[kk].q[1] = *(const uint4*)(pr + ko + 16);
            }
        }

        // --- O += P * V ---
#pragma unroll
        for (int t = 0; t < 4; ++t) {
#pragma unroll
            for (int kk = 0; kk < 2; ++kk) {
                const __bf16* vr = &Vt[((t << 4) + l16) * 64 + kk * 32 + (half << 3)];
                ABFrag bf;
                bf.q[0] = *(const uint4*)(vr);
                bf.q[1] = *(const uint4*)(vr + 16);
                o[t] = wmma_bf16(pa[kk].v, bf.v, o[t]);
            }
        }
    }

    // --- normalize and write y[b, q, h*64 + d] as bf16 ---
#pragma unroll
    for (int t = 0; t < 4; ++t)
#pragma unroll
        for (int e = 0; e < 8; ++e) {
            const float val = o[t][e] / lrow[e];
            const int r = q0 + e + (half << 3);
            const int d = (t << 4) + l16;
            Y[(((size_t)b * NQ + r) << 9) + (h << 6) + d] = (__bf16)val;
        }
}

// ---------------------------------------------------------------------------
// Host launcher
// ---------------------------------------------------------------------------
extern "C" void kernel_launch(void* const* d_in, const int* in_sizes, int n_in,
                              void* d_out, int out_size, void* d_ws, size_t ws_size,
                              hipStream_t stream) {
    const float* x_ctx    = (const float*)d_in[0];
    const float* x_query  = (const float*)d_in[1];
    const float* pos_ctx  = (const float*)d_in[2];
    const float* pos_qry  = (const float*)d_in[3];
    const float* Wq = (const float*)d_in[4];
    const float* bq = (const float*)d_in[5];
    const float* Wk = (const float*)d_in[6];
    const float* bk = (const float*)d_in[7];
    const float* Wv = (const float*)d_in[8];
    const float* bv = (const float*)d_in[9];
    const float* Wo = (const float*)d_in[10];
    const float* bo = (const float*)d_in[11];
    float* out = (float*)d_out;

    // workspace sub-allocation (256B aligned)
    char* ws = (char*)d_ws;
    size_t off = 0;
    auto alloc = [&](size_t elems) {
        void* p = ws + off;
        off += (elems * sizeof(__bf16) + 255) & ~(size_t)255;
        return (__bf16*)p;
    };
    __bf16* xqb = alloc((size_t)BATCH * NQ * CEMB);
    __bf16* xcb = alloc((size_t)BATCH * NC * CEMB);
    __bf16* Wqb = alloc((size_t)CEMB * CEMB);
    __bf16* Wkb = alloc((size_t)CEMB * CEMB);
    __bf16* Wvb = alloc((size_t)CEMB * CEMB);
    __bf16* Wob = alloc((size_t)CEMB * CEMB);
    __bf16* qb  = alloc((size_t)BATCH * NQ * CEMB);
    __bf16* kb_ = alloc((size_t)BATCH * NC * CEMB);
    __bf16* vb_ = alloc((size_t)BATCH * NC * CEMB);
    __bf16* yb  = alloc((size_t)BATCH * NQ * CEMB);

    auto cvt = [&](const float* src, __bf16* dst, int n) {
        int blocks = (n + 255) / 256;
        if (blocks > 4096) blocks = 4096;
        cvt_bf16_kernel<<<blocks, 256, 0, stream>>>(src, dst, n);
    };
    cvt(x_query, xqb, BATCH * NQ * CEMB);
    cvt(x_ctx,   xcb, BATCH * NC * CEMB);
    cvt(Wq, Wqb, CEMB * CEMB);
    cvt(Wk, Wkb, CEMB * CEMB);
    cvt(Wv, Wvb, CEMB * CEMB);
    cvt(Wo, Wob, CEMB * CEMB);

    // Q/K/V projections (q pre-scaled by 1/sqrt(64) = 0.125)
    proj_qkv_kernel<<<(BATCH * NQ) / 16, 256, 0, stream>>>(
        xqb, Wqb, bq, qb, BATCH * NQ, NQ, 0.125f);
    proj_qkv_kernel<<<(BATCH * NC) / 16, 256, 0, stream>>>(
        xcb, Wkb, bk, kb_, BATCH * NC, NC, 1.0f);
    proj_qkv_kernel<<<(BATCH * NC) / 16, 256, 0, stream>>>(
        xcb, Wvb, bv, vb_, BATCH * NC, NC, 1.0f);

    // Flash attention: B * H * (NQ/64) blocks
    attn_kernel<<<BATCH * NHEAD * (NQ / 64), 128, 0, stream>>>(
        qb, kb_, vb_, pos_ctx, pos_qry, yb);

    // Output projection -> fp32 d_out
    proj_out_kernel<<<(BATCH * NQ) / 16, 256, 0, stream>>>(
        yb, Wob, bo, out, BATCH * NQ);
}
